// CausalSelfAttention_11725260718275
// MI455X (gfx1250) — matrix operand order repaired and checked by reading, not compiled, not run on hardware
//
#include <hip/hip_runtime.h>

// ---------------------------------------------------------------------------
// Problem constants (B=2, T=2048, C=1024, H=16, D=64)
// ---------------------------------------------------------------------------
#define Bsz 2
#define Tsz 2048
#define Csz 1024
#define Hn  16
#define Dh  64

#define QSCL 0.18033688011112042f       /* 0.125 * log2(e) */
#define MASKV (-1.4426950408889634e10f) /* -1e10 * log2(e)  */

typedef unsigned short u16;
typedef __bf16 bf16;
typedef bf16  v16bf __attribute__((ext_vector_type(16)));
typedef float v8f   __attribute__((ext_vector_type(8)));
typedef u16   u16x8 __attribute__((ext_vector_type(8)));

union Op16 { v16bf v; u16x8 h[2]; u16 u[16]; };

static __device__ inline u16 f2bf(float f) {
  union { bf16 b; u16 u; } cv; cv.b = (bf16)f; return cv.u;
}
static __device__ inline float bf2f(u16 u) {
  union { float f; unsigned int i; } cv; cv.i = ((unsigned int)u) << 16; return cv.f;
}
static __device__ inline float fexp2(float x) { return __builtin_amdgcn_exp2f(x); }

// ---------------------------------------------------------------------------
// f32 -> bf16 (plain)
// ---------------------------------------------------------------------------
__global__ void cvt_f32_bf16(const float* __restrict__ in, u16* __restrict__ out, int n) {
  int i = blockIdx.x * blockDim.x + threadIdx.x;
  if (i < n) out[i] = f2bf(in[i]);
}

// ---------------------------------------------------------------------------
// f32 W[N,K] -> bf16 W^T[K,N]  (one-time; K = N = Csz = 1024)
// ---------------------------------------------------------------------------
__global__ void cvt_w_transpose(const float* __restrict__ w, u16* __restrict__ wt) {
  int idx = blockIdx.x * blockDim.x + threadIdx.x;   // < K*N
  int kk = idx & (Csz - 1);
  int n  = idx >> 10;
  wt[(size_t)kk * Csz + n] = f2bf(w[idx]);
}

// ---------------------------------------------------------------------------
// GEMM: Out[M,N] = A[M,K] @ W^T  with  WT[K,N] bf16 row-major, + bias[N]
// LDS-free: all WMMA operands are direct contiguous global loads.
// Block: 256 threads = 8 waves; tile 128x64; K step 32.
// ---------------------------------------------------------------------------
template <bool OUT_BF16>
__global__ __launch_bounds__(256)
void gemm_bf16(const u16* __restrict__ A, const u16* __restrict__ WT,
               const float* __restrict__ bias, void* __restrict__ Out,
               int M, int N, int K) {
  const int tid  = threadIdx.x;
  const int wave = tid >> 5;
  const int lane = tid & 31;
  const int half = lane >> 4;
  const int ln   = lane & 15;
  const int mbase = blockIdx.x * 128;
  const int nbase = blockIdx.y * 64;
  const int mrow  = mbase + wave * 16 + ln;

  v8f acc[4];
#pragma unroll
  for (int nt = 0; nt < 4; ++nt)
#pragma unroll
    for (int r = 0; r < 8; ++r) acc[nt][r] = 0.0f;

  const u16* arow = A + (size_t)mrow * K + half * 8;
  const u16* brow = WT + nbase + ln;

  for (int k0 = 0; k0 < K; k0 += 32) {
    Op16 a;
    a.h[0] = *(const u16x8*)(arow + k0);
    a.h[1] = *(const u16x8*)(arow + k0 + 16);
    __builtin_prefetch(arow + k0 + 32, 0, 3);
    __builtin_prefetch(brow + (size_t)(k0 + 32) * N, 0, 3);
#pragma unroll
    for (int nt = 0; nt < 4; ++nt) {
      Op16 b;
      const u16* bp = brow + (size_t)k0 * N + nt * 16;
#pragma unroll
      for (int s = 0; s < 16; ++s)
        b.u[s] = bp[(size_t)((s & 7) + 8 * half + 16 * (s >> 3)) * N];
      acc[nt] = __builtin_amdgcn_wmma_f32_16x16x32_bf16(
          false, a.v, false, b.v, (short)0, acc[nt], false, false);
    }
  }

#pragma unroll
  for (int nt = 0; nt < 4; ++nt) {
    float bv = bias[nbase + nt * 16 + ln];
#pragma unroll
    for (int r = 0; r < 8; ++r) {
      float v = acc[nt][r] + bv;
      size_t off = (size_t)(mbase + wave * 16 + r + 8 * half) * N + nbase + nt * 16 + ln;
      if (OUT_BF16) ((u16*)Out)[off] = f2bf(v);
      else          ((float*)Out)[off] = v;
    }
  }
}

// ---------------------------------------------------------------------------
// RoPE on Q,K + layout change; V pass-through.
//   Q -> qT [B,H,D,T] scaled by 0.125*log2(e)  (log2-domain softmax)
//   K -> kR [B,H,T,D]
//   V -> vT [B,H,D,T]
// theta_i = 10000^(2*(i-1)/D)  (faithful to reference)
// ---------------------------------------------------------------------------
__global__ void rope_split(const u16* __restrict__ qin, const u16* __restrict__ kin,
                           const u16* __restrict__ vin,
                           u16* __restrict__ qT, u16* __restrict__ kR,
                           u16* __restrict__ vT) {
  int idx = blockIdx.x * blockDim.x + threadIdx.x;  // pair index in [0, B*T*C/2)
  int which = blockIdx.y;
  const u16* src = (which == 0) ? qin : (which == 1) ? kin : vin;

  int c2 = idx & 511;
  int t  = (idx >> 9) & 2047;
  int b  = idx >> 20;
  int c  = c2 * 2;
  int h  = c >> 6;
  int d  = c & 63;

  float x0 = bf2f(src[(size_t)idx * 2]);
  float x1 = bf2f(src[(size_t)idx * 2 + 1]);

  if (which == 2) { // V: copy transposed
    size_t o = (((size_t)(b * Hn + h) * Dh + d) * Tsz + t);
    vT[o]       = src[(size_t)idx * 2];
    vT[o + Tsz] = src[(size_t)idx * 2 + 1];
    return;
  }

  int j = d >> 1;
  float theta = __powf(10000.0f, 2.0f * ((float)j - 1.0f) / 64.0f);
  float ang = (float)t * theta;
  float sn, cs;
  __sincosf(ang, &sn, &cs);
  float rr = x0 * cs - x1 * sn;
  float ri = x0 * sn + x1 * cs;

  if (which == 0) { // Q transposed + folded scale
    size_t o = (((size_t)(b * Hn + h) * Dh + d) * Tsz + t);
    qT[o]       = f2bf(rr * QSCL);
    qT[o + Tsz] = f2bf(ri * QSCL);
  } else {          // K row-major
    size_t o = (((size_t)(b * Hn + h) * Tsz + t) * Dh + d);
    kR[o]     = f2bf(rr);
    kR[o + 1] = f2bf(ri);
  }
}

// ---------------------------------------------------------------------------
// Flash attention (causal), transposed formulation, log2-domain softmax:
//   S^T = K_chunk @ Q^T   (lane = query, VGPR row = key)
//   O^T = V^T @ P^T       (lane = query, VGPR row = d)
// Wave owns 32 queries (two 16-query groups): K/V operand loads are shared
// across both groups (2x arithmetic intensity). Only the LAST key chunk is
// masked; interior chunks skip all compare/select work.
// ---------------------------------------------------------------------------
__global__ __launch_bounds__(256)
void flash_attn(const u16* __restrict__ qT, const u16* __restrict__ kR,
                const u16* __restrict__ vT, u16* __restrict__ yo) {
  __shared__ u16 pls_s[8][32 * 32];  // per-wave: P^T [key][32 queries] (2KB)

  const int tid  = threadIdx.x;
  const int wave = tid >> 5;
  const int lane = tid & 31;
  const int half = lane >> 4;
  const int ln   = lane & 15;
  const int bh   = blockIdx.y;                  // b*H + h
  const int q0   = blockIdx.x * 256 + wave * 32; // 32 queries per wave
  const size_t baseT = (size_t)bh * Dh * Tsz;   // qT / vT base
  const size_t baseK = (size_t)bh * Tsz * Dh;   // kR base
  u16* pls = pls_s[wave];

  // ---- preload Q^T B-operands: [group][d-chunk], lane = query, elements = d
  Op16 bq[2][2];
#pragma unroll
  for (int g = 0; g < 2; ++g)
#pragma unroll
    for (int gc = 0; gc < 2; ++gc) {
      const u16* p = qT + baseT + (size_t)(gc * 32) * Tsz + q0 + g * 16 + ln;
#pragma unroll
      for (int s = 0; s < 16; ++s)
        bq[g][gc].u[s] = p[(size_t)((s & 7) + 8 * half + 16 * (s >> 3)) * Tsz];
    }

  const int query[2] = { q0 + ln, q0 + 16 + ln };
  float mst[2] = { -3.0e38f, -3.0e38f }, lst[2] = { 0.0f, 0.0f };
  v8f acc[2][4];
#pragma unroll
  for (int g = 0; g < 2; ++g)
#pragma unroll
    for (int nt = 0; nt < 4; ++nt)
#pragma unroll
      for (int r = 0; r < 8; ++r) acc[g][nt][r] = 0.0f;

  auto chunk = [&](int jb, bool masked) {
    // ---- K A-operands, shared across both query groups
    Op16 ak[2][2]; // [key-tile][d-chunk]
#pragma unroll
    for (int t = 0; t < 2; ++t) {
      const u16* kr = kR + baseK + (size_t)(jb + t * 16 + ln) * Dh + half * 8;
      ak[t][0].h[0] = *(const u16x8*)(kr);
      ak[t][0].h[1] = *(const u16x8*)(kr + 16);
      ak[t][1].h[0] = *(const u16x8*)(kr + 32);
      ak[t][1].h[1] = *(const u16x8*)(kr + 48);
    }

    // ---- S^T tiles: [group][key-tile]
    v8f st[2][2];
#pragma unroll
    for (int g = 0; g < 2; ++g)
#pragma unroll
      for (int t = 0; t < 2; ++t) {
        v8f s;
#pragma unroll
        for (int r = 0; r < 8; ++r) s[r] = 0.0f;
        s = __builtin_amdgcn_wmma_f32_16x16x32_bf16(false, ak[t][0].v, false, bq[g][0].v, (short)0, s, false, false);
        s = __builtin_amdgcn_wmma_f32_16x16x32_bf16(false, ak[t][1].v, false, bq[g][1].v, (short)0, s, false, false);
        if (masked) {
#pragma unroll
          for (int r = 0; r < 8; ++r) {
            int key = jb + t * 16 + r + 8 * half;
            if (key > query[g]) s[r] = MASKV;
          }
        }
        st[g][t] = s;
      }

    // ---- online softmax per group (log2 domain)
#pragma unroll
    for (int g = 0; g < 2; ++g) {
      float cm = -3.0e38f;
#pragma unroll
      for (int t = 0; t < 2; ++t)
#pragma unroll
        for (int r = 0; r < 8; ++r) cm = fmaxf(cm, st[g][t][r]);
      cm = fmaxf(cm, __shfl_xor(cm, 16, 32));

      float mn   = fmaxf(mst[g], cm);
      float corr = fexp2(mst[g] - mn);
      mst[g] = mn;

      float rs = 0.0f;
#pragma unroll
      for (int t = 0; t < 2; ++t)
#pragma unroll
        for (int r = 0; r < 8; ++r) {
          float pv = fexp2(st[g][t][r] - mn);
          st[g][t][r] = pv;
          rs += pv;
        }
      rs += __shfl_xor(rs, 16, 32);
      lst[g] = lst[g] * corr + rs;

#pragma unroll
      for (int nt = 0; nt < 4; ++nt)
#pragma unroll
        for (int r = 0; r < 8; ++r) acc[g][nt][r] *= corr;

      // stage P^T: pls[key_local*32 + g*16 + query]
#pragma unroll
      for (int t = 0; t < 2; ++t)
#pragma unroll
        for (int r = 0; r < 8; ++r)
          pls[(t * 16 + r + 8 * half) * 32 + g * 16 + ln] = f2bf(st[g][t][r]);
    }

    // ---- P^T B-operands per group
    Op16 bp[2];
#pragma unroll
    for (int g = 0; g < 2; ++g) {
      const u16* pp = pls + g * 16 + ln;
#pragma unroll
      for (int s = 0; s < 16; ++s)
        bp[g].u[s] = pp[((s & 7) + 8 * half + 16 * (s >> 3)) * 32];
    }

    // ---- O^T += V^T @ P^T ; V^T A-operands shared across groups
#pragma unroll
    for (int nt = 0; nt < 4; ++nt) {
      Op16 av;
      const u16* vp = vT + baseT + (size_t)(nt * 16 + ln) * Tsz + jb + half * 8;
      av.h[0] = *(const u16x8*)(vp);
      av.h[1] = *(const u16x8*)(vp + 16);
#pragma unroll
      for (int g = 0; g < 2; ++g)
        acc[g][nt] = __builtin_amdgcn_wmma_f32_16x16x32_bf16(
            false, av.v, false, bp[g].v, (short)0, acc[g][nt], false, false);
    }
  };

  const int nchunks = (q0 >> 5) + 1;   // q0 is a multiple of 32
  for (int cch = 0; cch < nchunks - 1; ++cch) {
    const int jb = cch * 32;
    __builtin_prefetch(kR + baseK + (size_t)(jb + 32 + lane) * Dh, 0, 3);
    __builtin_prefetch(vT + baseT + (size_t)(16 + ln) * Tsz + jb + 32, 0, 3);
    chunk(jb, false);
  }
  chunk((nchunks - 1) * 32, true);     // only the last chunk needs the mask

  // ---- normalize + store y bf16 [B,T,C]; vectorized 16B stores
  const int b = bh >> 4, h = bh & 15;
#pragma unroll
  for (int g = 0; g < 2; ++g) {
    float inv = 1.0f / lst[g];
    size_t rowo = ((size_t)(b * Tsz + query[g])) * Csz + h * Dh + half * 8;
#pragma unroll
    for (int nt = 0; nt < 4; ++nt) {
      u16x8 w;
#pragma unroll
      for (int r = 0; r < 8; ++r) w[r] = f2bf(acc[g][nt][r] * inv);
      *(u16x8*)(yo + rowo + nt * 16) = w;   // d = nt*16 + 8*half + r
    }
  }
}

// ---------------------------------------------------------------------------
// Launch
// ---------------------------------------------------------------------------
extern "C" void kernel_launch(void* const* d_in, const int* in_sizes, int n_in,
                              void* d_out, int out_size, void* d_ws, size_t ws_size,
                              hipStream_t stream) {
  (void)in_sizes; (void)n_in; (void)out_size; (void)ws_size;
  const float* x  = (const float*)d_in[0];
  const float* Wq = (const float*)d_in[1];
  const float* bq = (const float*)d_in[2];
  const float* Wk = (const float*)d_in[3];
  const float* bk = (const float*)d_in[4];
  const float* Wv = (const float*)d_in[5];
  const float* bv = (const float*)d_in[6];
  const float* Wp = (const float*)d_in[7];
  const float* bp = (const float*)d_in[8];
  float* out = (float*)d_out;

  const size_t NX = (size_t)Bsz * Tsz * Csz;  // 4,194,304
  const size_t NW = (size_t)Csz * Csz;        // 1,048,576

  char* ws = (char*)d_ws;
  u16* xb   = (u16*)ws; ws += NX * 2;
  u16* wqt  = (u16*)ws; ws += NW * 2;   // W^T [K,N] bf16
  u16* wkt  = (u16*)ws; ws += NW * 2;
  u16* wvt  = (u16*)ws; ws += NW * 2;
  u16* wpt  = (u16*)ws; ws += NW * 2;
  u16* qlin = (u16*)ws; ws += NX * 2;
  u16* klin = (u16*)ws; ws += NX * 2;
  u16* vlin = (u16*)ws; ws += NX * 2;
  u16* qTb  = (u16*)ws; ws += NX * 2;   // [B,H,D,T] (scaled)
  u16* kRb  = (u16*)ws; ws += NX * 2;   // [B,H,T,D]
  u16* vTb  = (u16*)ws; ws += NX * 2;   // [B,H,D,T]
  u16* yb   = (u16*)ws; ws += NX * 2;   // total ~72 MB

  cvt_f32_bf16   <<<dim3((unsigned)((NX + 255) / 256)), 256, 0, stream>>>(x,  xb,  (int)NX);
  cvt_w_transpose<<<dim3((unsigned)((NW + 255) / 256)), 256, 0, stream>>>(Wq, wqt);
  cvt_w_transpose<<<dim3((unsigned)((NW + 255) / 256)), 256, 0, stream>>>(Wk, wkt);
  cvt_w_transpose<<<dim3((unsigned)((NW + 255) / 256)), 256, 0, stream>>>(Wv, wvt);
  cvt_w_transpose<<<dim3((unsigned)((NW + 255) / 256)), 256, 0, stream>>>(Wp, wpt);

  dim3 gg(Bsz * Tsz / 128, Csz / 64);   // 32 x 16
  gemm_bf16<true><<<gg, 256, 0, stream>>>(xb, wqt, bq, qlin, Bsz * Tsz, Csz, Csz);
  gemm_bf16<true><<<gg, 256, 0, stream>>>(xb, wkt, bk, klin, Bsz * Tsz, Csz, Csz);
  gemm_bf16<true><<<gg, 256, 0, stream>>>(xb, wvt, bv, vlin, Bsz * Tsz, Csz, Csz);

  rope_split<<<dim3((unsigned)(NX / 2 / 256), 3), 256, 0, stream>>>(qlin, klin, vlin, qTb, kRb, vTb);

  flash_attn<<<dim3(Tsz / 256, Bsz * Hn), 256, 0, stream>>>(qTb, kRb, vTb, yb);

  gemm_bf16<false><<<gg, 256, 0, stream>>>(yb, wpt, bp, out, Bsz * Tsz, Csz, Csz);
}
